// EU2_Net_78357383348641
// MI455X (gfx1250) — matrix operand order repaired and checked by reading, not compiled, not run on hardware
//
#include <hip/hip_runtime.h>
#include <hip/hip_bf16.h>

// ---------------------------------------------------------------------------
// EU2-Net forward on MI455X (gfx1250).
// Internal format: f16 NHWC, channels padded to power-of-two (>=16).
// Conv = implicit GEMM on V_WMMA_F32_16X16X32_F16, 16Mx64N per wave,
// contiguous b128 fragment loads, f32 accumulation, fused BN+ReLU epilogue.
// Waves within a block share the N-tile (pixel columns) so the heavy B
// operand gets WGP$ reuse; global_prefetch_b8 covers the next K-step.
// ---------------------------------------------------------------------------

typedef __attribute__((ext_vector_type(16))) _Float16 v16h;
typedef __attribute__((ext_vector_type(8)))  _Float16 v8h;
typedef __attribute__((ext_vector_type(8)))  float    v8f;

#define INV_BN 0.9999950000374997f   // 1/sqrt(1+1e-5)

// --- fold BN (eval) + conv bias into per-channel scale/shift ----------------
__global__ void fold_bn_kernel(const float* g, const float* b, const float* bb,
                               float* scale, float* shift, int n) {
    int i = blockIdx.x * blockDim.x + threadIdx.x;
    if (i >= n) return;
    float s  = g  ? (INV_BN * g[i]) : 1.0f;
    float sh = (bb ? bb[i] * s : 0.0f) + (b ? b[i] : 0.0f);
    scale[i] = s;
    shift[i] = sh;
}

// --- weight prep: OIHW f32 -> f16 [Mpad][Ktp], K = (kh*k+kw)*Cpad + ci ------
__global__ void wprep_kernel(const float* __restrict__ w, _Float16* __restrict__ w16,
                             int Cout, int Cin, int Cpad, int ksz, int Ktp, int Mpad) {
    long i = (long)blockIdx.x * blockDim.x + threadIdx.x;
    long total = (long)Mpad * Ktp;
    if (i >= total) return;
    int kk = (int)(i % Ktp);
    int m  = (int)(i / Ktp);
    int sp = kk / Cpad;
    int ci = kk - sp * Cpad;
    int k2 = ksz * ksz;
    float v = 0.0f;
    if (m < Cout && ci < Cin && sp < k2) {
        int kh = sp / ksz, kw = sp - (sp / ksz) * ksz;
        v = w[(((long)m * Cin + ci) * ksz + kh) * ksz + kw];
    }
    w16[i] = (_Float16)v;
}

// --- input convert: f32 NCHW (C=3) -> f16 NHWC (Cpad, zero padded) ----------
__global__ void cvt_in_kernel(const float* __restrict__ x, _Float16* __restrict__ y,
                              int Bn, int C, int Cpad, int HW) {
    long i = (long)blockIdx.x * blockDim.x + threadIdx.x;
    long total = (long)Bn * HW * Cpad;
    if (i >= total) return;
    int c = (int)(i % Cpad);
    long t = i / Cpad;
    int p = (int)(t % HW);
    int b = (int)(t / HW);
    float v = (c < C) ? x[((long)b * C + c) * HW + p] : 0.0f;
    y[i] = (_Float16)v;
}

// --- implicit GEMM conv, f16 NHWC in, f16 NHWC out (or f32 NCHW final) ------
__global__ void __launch_bounds__(256)
conv_wmma_kernel(const _Float16* __restrict__ xin, const _Float16* __restrict__ w16,
                 const float* __restrict__ scale, const float* __restrict__ shift,
                 void* __restrict__ yout,
                 int Bn, int H, int W, int shC /*log2(Cpad)*/, int Cout,
                 int ksz, int pad, int dil, int Ktp, int relu, int store_nchw_f32) {
    const int lane = threadIdx.x & 31;
    const int wid  = threadIdx.x >> 5;
    const int grp  = lane >> 4;
    const int lid  = lane & 15;
    const int Cpad = 1 << shC;
    const int cmask = Cpad - 1;
    const int k2 = ksz * ksz;

    const long N    = (long)Bn * H * W;
    const long Nt64 = (N + 63) >> 6;
    const int  Mtiles = (Cout + 15) >> 4;

    long tile = (long)blockIdx.x * (blockDim.x >> 5) + wid;
    if (tile >= Nt64 * (long)Mtiles) return;        // wave-uniform exit: EXEC stays full
    // Waves in a block share nt (B columns -> WGP$ reuse) and span mt.
    const int  mt = (int)(tile % Mtiles);
    const long nt = tile / Mtiles;

    // Decode the 4 pixel columns handled by this lane (N-tiles t=0..3).
    long nc[4]; int ph[4], pw[4], pbv[4]; bool nv[4]; const _Float16* xb[4];
#pragma unroll
    for (int t = 0; t < 4; ++t) {
        nc[t] = nt * 64 + t * 16 + lid;
        nv[t] = (nc[t] < N);
        long cl = nv[t] ? nc[t] : (N - 1);
        pbv[t] = (int)(cl / ((long)H * W));
        int r = (int)(cl % ((long)H * W));
        ph[t] = r / W;
        pw[t] = r - (r / W) * W;
        xb[t] = xin + (cl << shC);                  // NHWC pixel base
    }

    const _Float16* wrow = w16 + (long)(mt * 16 + lid) * Ktp;

    v8f acc0 = {}, acc1 = {}, acc2 = {}, acc3 = {};

    for (int k0 = 0; k0 < Ktp; k0 += 32) {
        // ---- A fragment: two contiguous b128 loads per lane (ISA 7.12.2) ----
        const _Float16* ap = wrow + k0 + grp * 16;
        v8h a0 = *(const v8h*)ap;
        v8h a1 = *(const v8h*)(ap + 8);
        v16h a = __builtin_shufflevector(a0, a1, 0,1,2,3,4,5,6,7,8,9,10,11,12,13,14,15);
        __builtin_prefetch(ap + 32, 0, 0);          // next K-step's A (row-contiguous)

        // ---- shared B-fragment geometry for this K block ----
        int Kb = k0 + grp * 16;
        int sp = Kb >> shC;                 // spatial offset index
        int cb = Kb & cmask;                // channel base (multiple of 16)
        int kh = sp / ksz;
        int kw = sp - kh * ksz;
        int dh = kh * dil - pad;
        int dw = kw * dil - pad;
        bool sv = (sp < k2);
        long off = (((long)dh * W + dw) << shC) + cb;

        v16h bfr;
#define LOAD_B(t)                                                              \
        bfr = (v16h)(_Float16)0;                                               \
        if (sv && nv[t] && (unsigned)(ph[t] + dh) < (unsigned)H &&             \
            (unsigned)(pw[t] + dw) < (unsigned)W) {                            \
            const _Float16* p = xb[t] + off;                                   \
            v8h b0 = *(const v8h*)p;                                           \
            v8h b1 = *(const v8h*)(p + 8);                                     \
            __builtin_prefetch(p + 32, 0, 0);                                  \
            bfr = __builtin_shufflevector(b0, b1, 0,1,2,3,4,5,6,7,8,9,10,11,12,13,14,15); \
        }

        LOAD_B(0)
        acc0 = __builtin_amdgcn_wmma_f32_16x16x32_f16(false, a, false, bfr, (short)0, acc0, false, false);
        LOAD_B(1)
        acc1 = __builtin_amdgcn_wmma_f32_16x16x32_f16(false, a, false, bfr, (short)0, acc1, false, false);
        LOAD_B(2)
        acc2 = __builtin_amdgcn_wmma_f32_16x16x32_f16(false, a, false, bfr, (short)0, acc2, false, false);
        LOAD_B(3)
        acc3 = __builtin_amdgcn_wmma_f32_16x16x32_f16(false, a, false, bfr, (short)0, acc3, false, false);
#undef LOAD_B
    }

    // ---- epilogue: C/D layout VGPR r -> M = mt*16 + grp*8 + r --------------
    const int mbase = mt * 16 + grp * 8;
    if (!store_nchw_f32) {
        _Float16* yb = (_Float16*)yout;
#define STORE_T(t, accv)                                                       \
        if (nv[t]) {                                                           \
            v8h o;                                                             \
            _Pragma("unroll")                                                  \
            for (int r = 0; r < 8; ++r) {                                      \
                float v = accv[r] * scale[mbase + r] + shift[mbase + r];       \
                if (relu) v = fmaxf(v, 0.0f);                                  \
                o[r] = (_Float16)v;                                            \
            }                                                                  \
            *(v8h*)(yb + nc[t] * Cout + mbase) = o;                            \
        }
        STORE_T(0, acc0)
        STORE_T(1, acc1)
        STORE_T(2, acc2)
        STORE_T(3, acc3)
#undef STORE_T
    } else {
        float* yb = (float*)yout;
#pragma unroll
        for (int r = 0; r < 8; ++r) {
            int m = mbase + r;
            if (m < Cout) {
                float v0 = acc0[r], v1 = acc1[r], v2 = acc2[r], v3 = acc3[r];
                float s = scale[m], b = shift[m];
                v0 = v0 * s + b; v1 = v1 * s + b; v2 = v2 * s + b; v3 = v3 * s + b;
                if (relu) { v0 = fmaxf(v0, 0.f); v1 = fmaxf(v1, 0.f); v2 = fmaxf(v2, 0.f); v3 = fmaxf(v3, 0.f); }
                if (nv[0]) yb[(((long)pbv[0] * Cout + m) * H + ph[0]) * W + pw[0]] = v0;
                if (nv[1]) yb[(((long)pbv[1] * Cout + m) * H + ph[1]) * W + pw[1]] = v1;
                if (nv[2]) yb[(((long)pbv[2] * Cout + m) * H + ph[2]) * W + pw[2]] = v2;
                if (nv[3]) yb[(((long)pbv[3] * Cout + m) * H + ph[3]) * W + pw[3]] = v3;
            }
        }
    }
}

// --- glue kernels (f16 NHWC) ------------------------------------------------
__global__ void maxpool2_kernel(const _Float16* __restrict__ x, _Float16* __restrict__ y,
                                int Bn, int Ho, int Wo, int C) {
    long i = (long)blockIdx.x * blockDim.x + threadIdx.x;
    long total = (long)Bn * Ho * Wo * C;
    if (i >= total) return;
    int c = (int)(i % C); long t = i / C;
    int wo = (int)(t % Wo); t /= Wo;
    int ho = (int)(t % Ho); int b = (int)(t / Ho);
    int Wi = 2 * Wo;
    const _Float16* p = x + (((long)b * (2 * Ho) + 2 * ho) * Wi + 2 * wo) * C + c;
    float m0 = fmaxf((float)p[0], (float)p[C]);
    float m1 = fmaxf((float)p[(long)Wi * C], (float)p[(long)(Wi + 1) * C]);
    y[i] = (_Float16)fmaxf(m0, m1);
}

__global__ void concat_kernel(const _Float16* __restrict__ src, _Float16* __restrict__ dst,
                              long npix, int Cs, int Ctot, int coff) {
    long i = (long)blockIdx.x * blockDim.x + threadIdx.x;
    long total = npix * Cs;
    if (i >= total) return;
    int c = (int)(i % Cs);
    long p = i / Cs;
    dst[p * Ctot + coff + c] = src[i];
}

__global__ void upsample2_kernel(const _Float16* __restrict__ src, _Float16* __restrict__ dst,
                                 int Bn, int Hi, int Wi, int C) {
    long i = (long)blockIdx.x * blockDim.x + threadIdx.x;
    long total = (long)Bn * 4 * Hi * Wi * C;
    if (i >= total) return;
    int c = (int)(i % C); long t = i / C;
    int Wo = 2 * Wi, Ho = 2 * Hi;
    int wo = (int)(t % Wo); t /= Wo;
    int ho = (int)(t % Ho); int b = (int)(t / Ho);
    dst[i] = src[(((long)b * Hi + (ho >> 1)) * Wi + (wo >> 1)) * C + c];
}

// ASPP interleave (branches packed, stride br): out[b,2i+a,2j+q,c]=branch(2a+q)
__global__ void interleave4_kernel(const _Float16* __restrict__ src, _Float16* __restrict__ dst,
                                   int Bn, int h, int w, int C) {
    long br = (long)Bn * h * w * C;
    long i = (long)blockIdx.x * blockDim.x + threadIdx.x;
    long total = br * 4;
    if (i >= total) return;
    int c = (int)(i % C); long t = i / C;
    int Wo = 2 * w, Ho = 2 * h;
    int wo = (int)(t % Wo); t /= Wo;
    int ho = (int)(t % Ho); int b = (int)(t / Ho);
    int q = ((ho & 1) << 1) | (wo & 1);
    dst[i] = src[q * br + (((long)b * h + (ho >> 1)) * w + (wo >> 1)) * C + c];
}

// ---------------------------------------------------------------------------
static inline int nb(long n) { return (int)((n + 255) / 256); }
static inline int ilog2(int v) { int s = 0; while ((1 << s) < v) ++s; return s; }

static void launch_conv(hipStream_t s, const _Float16* x, const float* w,
                        const float* g, const float* bcoef, const float* bb,
                        float* scale, float* shift, _Float16* wbuf, void* y,
                        int Bn, int H, int W, int Cin, int Cpad, int Cout,
                        int ksz, int pad, int dil, int relu, int store_nchw) {
    int k2   = ksz * ksz;
    int Ktp  = ((k2 * Cpad + 31) / 32) * 32;
    int Mpad = ((Cout + 15) / 16) * 16;
    fold_bn_kernel<<<(Cout + 255) / 256, 256, 0, s>>>(g, bcoef, bb, scale, shift, Cout);
    wprep_kernel<<<nb((long)Mpad * Ktp), 256, 0, s>>>(w, wbuf, Cout, Cin, Cpad, ksz, Ktp, Mpad);
    long N     = (long)Bn * H * W;
    long waves = ((N + 63) >> 6) * (long)((Cout + 15) >> 4);
    int blocks = (int)((waves + 7) / 8);
    conv_wmma_kernel<<<blocks, 256, 0, s>>>(x, wbuf, scale, shift, y,
                                            Bn, H, W, ilog2(Cpad), Cout,
                                            ksz, pad, dil, Ktp, relu, store_nchw);
}

extern "C" void kernel_launch(void* const* d_in, const int* in_sizes, int n_in,
                              void* d_out, int out_size, void* d_ws, size_t ws_size,
                              hipStream_t stream) {
    (void)in_sizes; (void)n_in; (void)out_size; (void)ws_size;
    auto F = [&](int i) { return (const float*)d_in[i]; };

    // Leaf order: x, then params pytree with sorted dict keys at each level.
    // aspp: +0..3=b0..b3,+4=bp,+5..8=g0..g3,+9=gp,+10=w0,+11..13=w1..w3,+14=wp
    // cb:   +0=b1,+1=b2,+2=bb1,+3=bb2,+4=g1,+5=g2,+6=w1,+7=w2
    const int A3 = 1, A4 = 16, C1 = 31, C2 = 39, C3 = 47, C4 = 55;
    const int OUT_BB = 63, OUT_W = 64, U2 = 65, U3 = 73, U4 = 81;
    const int UP2_B = 89, UP2_BB = 90, UP2_G = 91, UP2_W = 92;

    // ---- workspace arena (bytes, 256-aligned); activations are f16 NHWC ----
    char* ws = (char*)d_ws;
    size_t off = 0;
    auto alloc = [&](size_t bytes) { size_t o = off; off = (off + bytes + 255) & ~(size_t)255; return o; };
    _Float16* e1   = (_Float16*)(ws + alloc(16777216));  // 4*256*256*32 f16
    _Float16* e2   = (_Float16*)(ws + alloc(8388608));   // 4*128*128*64
    _Float16* e3   = (_Float16*)(ws + alloc(4194304));   // 4*64*64*128
    _Float16* tA   = (_Float16*)(ws + alloc(33554432));
    _Float16* tB   = (_Float16*)(ws + alloc(33554432));
    _Float16* tC   = (_Float16*)(ws + alloc(33554432));
    _Float16* wbuf = (_Float16*)(ws + alloc(2359296));   // max prepped weights
    float*    sc   = (float*)(ws + alloc(1024));
    float*    sh   = (float*)(ws + alloc(1024));

    const int B = 4;
    hipStream_t s = stream;

    // ---- input: f32 NCHW C=3 -> f16 NHWC C=16 ----
    cvt_in_kernel<<<nb((long)B*256*256*16), 256, 0, s>>>(F(0), tA, B, 3, 16, 256*256);

    // ---- encoder ----
    launch_conv(s, tA, F(C1+6), F(C1+4), F(C1+0), F(C1+2), sc, sh, wbuf, tB, B, 256, 256, 3, 16, 32, 3, 1, 1, 1, 0);
    launch_conv(s, tB, F(C1+7), F(C1+5), F(C1+1), F(C1+3), sc, sh, wbuf, e1, B, 256, 256, 32, 32, 32, 3, 1, 1, 1, 0);
    maxpool2_kernel<<<nb((long)B*128*128*32), 256, 0, s>>>(e1, tA, B, 128, 128, 32);
    launch_conv(s, tA, F(C2+6), F(C2+4), F(C2+0), F(C2+2), sc, sh, wbuf, tB, B, 128, 128, 32, 32, 64, 3, 1, 1, 1, 0);
    launch_conv(s, tB, F(C2+7), F(C2+5), F(C2+1), F(C2+3), sc, sh, wbuf, e2, B, 128, 128, 64, 64, 64, 3, 1, 1, 1, 0);
    maxpool2_kernel<<<nb((long)B*64*64*64), 256, 0, s>>>(e2, tA, B, 64, 64, 64);
    launch_conv(s, tA, F(C3+6), F(C3+4), F(C3+0), F(C3+2), sc, sh, wbuf, tB, B, 64, 64, 64, 64, 128, 3, 1, 1, 1, 0);
    launch_conv(s, tB, F(C3+7), F(C3+5), F(C3+1), F(C3+3), sc, sh, wbuf, e3, B, 64, 64, 128, 128, 128, 3, 1, 1, 1, 0);
    maxpool2_kernel<<<nb((long)B*32*32*128), 256, 0, s>>>(e3, tA, B, 32, 32, 128);
    launch_conv(s, tA, F(C4+6), F(C4+4), F(C4+0), F(C4+2), sc, sh, wbuf, tB, B, 32, 32, 128, 128, 256, 3, 1, 1, 1, 0);
    launch_conv(s, tB, F(C4+7), F(C4+5), F(C4+1), F(C4+3), sc, sh, wbuf, tA, B, 32, 32, 256, 256, 256, 3, 1, 1, 1, 0);

    // ---- a4: ASPP on tA (256ch @32), oc=128 ----
    {
        long br = (long)B * 32 * 32 * 128;
        launch_conv(s, tA, F(A4+10), F(A4+5), F(A4+0), nullptr, sc, sh, wbuf, tB + 0*br, B, 32, 32, 256, 256, 128, 1, 0,  1, 1, 0);
        launch_conv(s, tA, F(A4+11), F(A4+6), F(A4+1), nullptr, sc, sh, wbuf, tB + 1*br, B, 32, 32, 256, 256, 128, 3, 6,  6, 1, 0);
        launch_conv(s, tA, F(A4+12), F(A4+7), F(A4+2), nullptr, sc, sh, wbuf, tB + 2*br, B, 32, 32, 256, 256, 128, 3, 12, 12, 1, 0);
        launch_conv(s, tA, F(A4+13), F(A4+8), F(A4+3), nullptr, sc, sh, wbuf, tB + 3*br, B, 32, 32, 256, 256, 128, 3, 18, 18, 1, 0);
        interleave4_kernel<<<nb(br * 4), 256, 0, s>>>(tB, tC, B, 32, 32, 128);
        launch_conv(s, tC, F(A4+14), F(A4+9), F(A4+4), nullptr, sc, sh, wbuf, tB, B, 64, 64, 128, 128, 128, 1, 0, 1, 1, 0); // d4 -> tB
    }
    concat_kernel<<<nb((long)B*64*64*128), 256, 0, s>>>(e3, tC, (long)B*64*64, 128, 256, 0);
    concat_kernel<<<nb((long)B*64*64*128), 256, 0, s>>>(tB, tC, (long)B*64*64, 128, 256, 128);
    launch_conv(s, tC, F(U4+6), F(U4+4), F(U4+0), F(U4+2), sc, sh, wbuf, tA, B, 64, 64, 256, 256, 128, 3, 1, 1, 1, 0);
    launch_conv(s, tA, F(U4+7), F(U4+5), F(U4+1), F(U4+3), sc, sh, wbuf, tB, B, 64, 64, 128, 128, 128, 3, 1, 1, 1, 0);

    // ---- a3: ASPP on tB (128ch @64), oc=64 ----
    {
        long br = (long)B * 64 * 64 * 64;
        launch_conv(s, tB, F(A3+10), F(A3+5), F(A3+0), nullptr, sc, sh, wbuf, tA + 0*br, B, 64, 64, 128, 128, 64, 1, 0,  1, 1, 0);
        launch_conv(s, tB, F(A3+11), F(A3+6), F(A3+1), nullptr, sc, sh, wbuf, tA + 1*br, B, 64, 64, 128, 128, 64, 3, 6,  6, 1, 0);
        launch_conv(s, tB, F(A3+12), F(A3+7), F(A3+2), nullptr, sc, sh, wbuf, tA + 2*br, B, 64, 64, 128, 128, 64, 3, 12, 12, 1, 0);
        launch_conv(s, tB, F(A3+13), F(A3+8), F(A3+3), nullptr, sc, sh, wbuf, tA + 3*br, B, 64, 64, 128, 128, 64, 3, 18, 18, 1, 0);
        interleave4_kernel<<<nb(br * 4), 256, 0, s>>>(tA, tC, B, 64, 64, 64);
        launch_conv(s, tC, F(A3+14), F(A3+9), F(A3+4), nullptr, sc, sh, wbuf, tA, B, 128, 128, 64, 64, 64, 1, 0, 1, 1, 0); // d3pre -> tA
    }
    concat_kernel<<<nb((long)B*128*128*64), 256, 0, s>>>(e2, tB, (long)B*128*128, 64, 128, 0);
    concat_kernel<<<nb((long)B*128*128*64), 256, 0, s>>>(tA, tB, (long)B*128*128, 64, 128, 64);
    launch_conv(s, tB, F(U3+6), F(U3+4), F(U3+0), F(U3+2), sc, sh, wbuf, tC, B, 128, 128, 128, 128, 64, 3, 1, 1, 1, 0);
    launch_conv(s, tC, F(U3+7), F(U3+5), F(U3+1), F(U3+3), sc, sh, wbuf, tA, B, 128, 128, 64, 64, 64, 3, 1, 1, 1, 0);

    // ---- d2: nearest 2x upsample + up2 conv ----
    upsample2_kernel<<<nb((long)B*256*256*64), 256, 0, s>>>(tA, tB, B, 128, 128, 64);
    launch_conv(s, tB, F(UP2_W), F(UP2_G), F(UP2_B), F(UP2_BB), sc, sh, wbuf, tC, B, 256, 256, 64, 64, 32, 3, 1, 1, 1, 0);
    concat_kernel<<<nb((long)B*256*256*32), 256, 0, s>>>(e1, tB, (long)B*256*256, 32, 64, 0);
    concat_kernel<<<nb((long)B*256*256*32), 256, 0, s>>>(tC, tB, (long)B*256*256, 32, 64, 32);
    launch_conv(s, tB, F(U2+6), F(U2+4), F(U2+0), F(U2+2), sc, sh, wbuf, tC, B, 256, 256, 64, 64, 16, 3, 1, 1, 1, 0);
    launch_conv(s, tC, F(U2+7), F(U2+5), F(U2+1), F(U2+3), sc, sh, wbuf, tA, B, 256, 256, 16, 16, 16, 3, 1, 1, 1, 0);

    // ---- out: 1x1 conv 16->1, bias only, no ReLU, f32 NCHW to d_out ----
    launch_conv(s, tA, F(OUT_W), nullptr, nullptr, F(OUT_BB), sc, sh, wbuf, d_out,
                B, 256, 256, 16, 16, 1, 1, 0, 1, 0, 1);
}
